// QuantizedLinear_618475291177
// MI455X (gfx1250) — compile-verified, matrix operand order checked by reading, not run on hardware
//
#include <hip/hip_runtime.h>

// ---------------------------------------------------------------------------
// QuantizedLinear for MI455X (gfx1250, wave32):
//   y[t,o] = scale * sum_k x[t,k]*(q[o,k]-zp) + bias[o]
// bf16 WMMA (v_wmma_f32_16x16x32_bf16), f32 accumulate.
// (q-zp) in [-128,127] is exact in bf16; scale folded into epilogue.
// Block 128x128 (8 waves, each 64x32), K staged 32-wide, double-buffered LDS.
// ---------------------------------------------------------------------------

typedef __attribute__((ext_vector_type(16))) __bf16 v16bf;
typedef __attribute__((ext_vector_type(8)))  __bf16 v8bf;
typedef __attribute__((ext_vector_type(8)))  float  v8f;

#define TOKENS 8192
#define IN_F   4096
#define OUT_F  4096

#define BM 128              // block tile M (tokens)
#define BN 128              // block tile N (out features)
#define BK 32               // K per stage (= WMMA K)
#define LSTR (BK + 8)       // LDS row stride in bf16 (80B: 16B-aligned, conflict pad)

__device__ __forceinline__ v16bf ld_frag(const __bf16* p) {
    // elements 0..7  : K = g*8 + 0..7        (at p)
    // elements 8..15 : K = 16 + g*8 + 0..7   (at p+16)
    v8bf lo = *(const v8bf*)p;
    v8bf hi = *(const v8bf*)(p + 16);
    v16bf f;
#pragma unroll
    for (int i = 0; i < 8; ++i) { f[i] = lo[i]; f[i + 8] = hi[i]; }
    return f;
}

__device__ __forceinline__ void cvt_store_a8(__bf16* dst, float4 lo, float4 hi) {
    v8bf t;
    t[0] = (__bf16)lo.x; t[1] = (__bf16)lo.y; t[2] = (__bf16)lo.z; t[3] = (__bf16)lo.w;
    t[4] = (__bf16)hi.x; t[5] = (__bf16)hi.y; t[6] = (__bf16)hi.z; t[7] = (__bf16)hi.w;
    *(v8bf*)dst = t;   // ds_store_b128
}

__device__ __forceinline__ void cvt_store_b8(__bf16* dst, int4 lo, int4 hi, int zp) {
    v8bf t;
    t[0] = (__bf16)(float)(lo.x - zp); t[1] = (__bf16)(float)(lo.y - zp);
    t[2] = (__bf16)(float)(lo.z - zp); t[3] = (__bf16)(float)(lo.w - zp);
    t[4] = (__bf16)(float)(hi.x - zp); t[5] = (__bf16)(float)(hi.y - zp);
    t[6] = (__bf16)(float)(hi.z - zp); t[7] = (__bf16)(float)(hi.w - zp);
    *(v8bf*)dst = t;   // ds_store_b128
}

__global__ __launch_bounds__(256)
void qlinear_wmma_kernel(const float* __restrict__ x,
                         const int*   __restrict__ qw,
                         const float* __restrict__ scale_p,
                         const int*   __restrict__ zp_p,
                         const float* __restrict__ bias,
                         float* __restrict__ out)
{
    __shared__ __bf16 As[2][BM * LSTR];   // x tile   (bf16)
    __shared__ __bf16 Bs[2][BN * LSTR];   // (q-zp)   (bf16)

    const int tid  = threadIdx.x;
    const int lane = tid & 31;
    const int wave = tid >> 5;            // 0..7
    const int g    = (lane >> 4) & 1;     // lane half (K split)
    const int ln   = lane & 15;           // row/col within fragment

    const int m0 = blockIdx.y * BM;
    const int n0 = blockIdx.x * BN;

    const int wm = (wave & 1) * 64;       // wave M offset: 0/64
    const int wn = (wave >> 1) * 32;      // wave N offset: 0/32/64/96

    const float scale = scale_p[0];
    const int   zp    = zp_p[0];

    // global->LDS staging: 128 rows x 32 cols of 4B elems, 256 threads.
    // Each thread owns 8 contiguous cols on rows r and r+64 -> ds_store_b128.
    const int r  = tid >> 2;              // 0..63
    const int c8 = (tid & 3) * 8;         // col base (8 elems)

    const float* apA = x  + (size_t)(m0 + r)      * IN_F + c8;
    const float* apB = x  + (size_t)(m0 + r + 64) * IN_F + c8;
    const int*   bpA = qw + (size_t)(n0 + r)      * IN_F + c8;
    const int*   bpB = qw + (size_t)(n0 + r + 64) * IN_F + c8;

    __bf16* sA0 = &As[0][0] + r * LSTR + c8;          // +64*LSTR for second row
    __bf16* sB0 = &Bs[0][0] + r * LSTR + c8;

    // fragment LDS read bases (per wave/lane), k-half offset g*8
    const int aoff = (wm + ln) * LSTR + g * 8;        // + 16*mf*LSTR
    const int boff = (wn + ln) * LSTR + g * 8;        // + 16*nf*LSTR

    v8f acc[4][2];
#pragma unroll
    for (int i = 0; i < 4; ++i)
#pragma unroll
        for (int j = 0; j < 2; ++j)
#pragma unroll
            for (int v = 0; v < 8; ++v)
                acc[i][j][v] = 0.0f;

    const int nstages = IN_F / BK;

    float4 a_reg[4];
    int4   b_reg[4];

    // ---------------- prologue: fetch + stage K-chunk 0 ----------------
    a_reg[0] = *(const float4*)(apA);     a_reg[1] = *(const float4*)(apA + 4);
    a_reg[2] = *(const float4*)(apB);     a_reg[3] = *(const float4*)(apB + 4);
    b_reg[0] = *(const int4*)  (bpA);     b_reg[1] = *(const int4*)  (bpA + 4);
    b_reg[2] = *(const int4*)  (bpB);     b_reg[3] = *(const int4*)  (bpB + 4);

    cvt_store_a8(sA0,             a_reg[0], a_reg[1]);
    cvt_store_a8(sA0 + 64 * LSTR, a_reg[2], a_reg[3]);
    cvt_store_b8(sB0,             b_reg[0], b_reg[1], zp);
    cvt_store_b8(sB0 + 64 * LSTR, b_reg[2], b_reg[3], zp);
    __syncthreads();

    // ---------------- main K loop, double buffered ----------------
    for (int s = 0; s < nstages; ++s) {
        const int buf = s & 1;

        // prefetch next stage into registers (hides HBM latency under WMMA)
        if (s + 1 < nstages) {
            const int k0 = (s + 1) * BK;
            a_reg[0] = *(const float4*)(apA + k0);  a_reg[1] = *(const float4*)(apA + k0 + 4);
            a_reg[2] = *(const float4*)(apB + k0);  a_reg[3] = *(const float4*)(apB + k0 + 4);
            b_reg[0] = *(const int4*)  (bpA + k0);  b_reg[1] = *(const int4*)  (bpA + k0 + 4);
            b_reg[2] = *(const int4*)  (bpB + k0);  b_reg[3] = *(const int4*)  (bpB + k0 + 4);
        }

        // load ALL fragments first, then the 8-WMMA chain (partial DS waits)
        {
            const __bf16* sa = &As[buf][0] + aoff;
            const __bf16* sb = &Bs[buf][0] + boff;

            v16bf bfr[2], afr[4];
#pragma unroll
            for (int nf = 0; nf < 2; ++nf)
                bfr[nf] = ld_frag(sb + 16 * nf * LSTR);
#pragma unroll
            for (int mf = 0; mf < 4; ++mf)
                afr[mf] = ld_frag(sa + 16 * mf * LSTR);

#pragma unroll
            for (int mf = 0; mf < 4; ++mf)
#pragma unroll
                for (int nf = 0; nf < 2; ++nf)
                    acc[mf][nf] = __builtin_amdgcn_wmma_f32_16x16x32_bf16(
                        /*neg_a=*/false, afr[mf],
                        /*neg_b=*/false, bfr[nf],
                        /*c_mod=*/(short)0, acc[mf][nf],
                        /*reuse_a=*/false, /*reuse_b=*/false);
        }

        // stage next chunk into the other buffer
        if (s + 1 < nstages) {
            __bf16* dA = sA0 + (buf ^ 1) * (BM * LSTR);
            __bf16* dB = sB0 + (buf ^ 1) * (BN * LSTR);
            cvt_store_a8(dA,             a_reg[0], a_reg[1]);
            cvt_store_a8(dA + 64 * LSTR, a_reg[2], a_reg[3]);
            cvt_store_b8(dB,             b_reg[0], b_reg[1], zp);
            cvt_store_b8(dB + 64 * LSTR, b_reg[2], b_reg[3], zp);
            __syncthreads();
        }
    }

    // ---------------- epilogue: y = scale*acc + bias ----------------
#pragma unroll
    for (int mf = 0; mf < 4; ++mf) {
#pragma unroll
        for (int nf = 0; nf < 2; ++nf) {
            const int col   = n0 + wn + 16 * nf + ln;
            const int rbase = m0 + wm + 16 * mf + g * 8;
            const float bcol = bias[col];
#pragma unroll
            for (int v = 0; v < 8; ++v)
                out[(size_t)(rbase + v) * OUT_F + col] = scale * acc[mf][nf][v] + bcol;
        }
    }
}

extern "C" void kernel_launch(void* const* d_in, const int* in_sizes, int n_in,
                              void* d_out, int out_size, void* d_ws, size_t ws_size,
                              hipStream_t stream) {
    (void)in_sizes; (void)n_in; (void)out_size; (void)d_ws; (void)ws_size;
    const float* x     = (const float*)d_in[0];
    const int*   qw    = (const int*)d_in[1];
    const float* scale = (const float*)d_in[2];
    const int*   zp    = (const int*)d_in[3];
    const float* bias  = (const float*)d_in[4];
    float* out = (float*)d_out;

    dim3 grid(OUT_F / BN, TOKENS / BM);   // 32 x 64 blocks
    dim3 block(256);                      // 8 waves (wave32)
    qlinear_wmma_kernel<<<grid, block, 0, stream>>>(x, qw, scale, zp, bias, out);
}
